// ChunkCrossAttention_70239895159099
// MI455X (gfx1250) — compile-verified, matrix operand order checked by reading, not run on hardware
//
#include <hip/hip_runtime.h>
#include <cstdint>

// ---------------------------------------------------------------------------
// CDNA5 (gfx1250) fused chunk cross-attention
//   K1: kv = X @ Wkv^T           (bf16 hi/lo split WMMA, f32 accum)
//   K2: flash attention          (double-buffered TDM K/V staging,
//                                 bf16 hi/lo scores, online softmax, P@V WMMA)
//   K3: out-proj + residual + LN (f32 VALU)
// ---------------------------------------------------------------------------

typedef __attribute__((ext_vector_type(16))) __bf16 v16bf;
typedef __attribute__((ext_vector_type(8)))  float  v8f;
typedef __attribute__((ext_vector_type(4)))  unsigned int v4u_t;
typedef __attribute__((ext_vector_type(8)))  int v8i_t;
typedef __attribute__((ext_vector_type(4)))  int v4i_t;

#if __has_builtin(__builtin_amdgcn_tensor_load_to_lds) && \
    __has_builtin(__builtin_amdgcn_s_wait_tensorcnt)
#define USE_TDM 1
#else
#define USE_TDM 0
#endif

__device__ __forceinline__ v8f wmma_bf16(v16bf a, v16bf b, v8f c) {
  // (neg_a, A, neg_b, B, c_mod, C, reuse_a, reuse_b)
  return __builtin_amdgcn_wmma_f32_16x16x32_bf16(false, a, false, b, (short)0, c,
                                                 false, false);
}

__device__ __forceinline__ v8f vzero8() {
  v8f z = {0.f, 0.f, 0.f, 0.f, 0.f, 0.f, 0.f, 0.f};
  return z;
}

// A/B fragment, row-major source. p points at &M[row][kchunk*32 + kbase].
// elems 0..7 <- k = kbase+0..7 ; elems 8..15 <- k = kbase+16..23
__device__ __forceinline__ v16bf frag_row(const __bf16* p) {
  v16bf f;
#pragma unroll
  for (int i = 0; i < 8; ++i) f[i] = p[i];
#pragma unroll
  for (int i = 0; i < 8; ++i) f[8 + i] = p[16 + i];
  return f;
}

// B fragment, column read (for V: rows = k, col = n). p = &M[kbase][col].
__device__ __forceinline__ v16bf frag_col(const __bf16* p, int ld) {
  v16bf f;
#pragma unroll
  for (int i = 0; i < 8; ++i) f[i] = p[i * ld];
#pragma unroll
  for (int i = 0; i < 8; ++i) f[8 + i] = p[(16 + i) * ld];
  return f;
}

#if USE_TDM
// 1-D TDM copy: ndw dwords from global gptr -> LDS at ldsptr.
// D# layout per CDNA5 ISA:
//   g0: [count=1|..], lds_addr, global_addr[31:0], global_addr[56:32]|type=2
//   g1: data_size=2(4B); tensor_dim0=ndw; tensor_dim1=1; tile_dim0=ndw;
//       tile_dim1=1; tensor_dim0_stride=ndw
__device__ __forceinline__ void tdm_load_1d(const void* ldsptr,
                                            const void* gptr, unsigned ndw) {
  unsigned long long ga = (unsigned long long)(uintptr_t)gptr;
  unsigned lds_off = (unsigned)(uintptr_t)ldsptr;  // generic ptr low 32 = LDS
  v4u_t g0;
  g0[0] = 1u;                                   // count=1, user descriptor
  g0[1] = lds_off;                              // LDS byte address
  g0[2] = (unsigned)(ga & 0xffffffffu);         // global addr lo
  g0[3] = (unsigned)((ga >> 32) & 0x1ffffffu) | (2u << 30);  // hi | type=2
  v8i_t g1;
  g1[0] = (int)(2u << 16);        // data_size = 4 bytes
  g1[1] = (int)(ndw << 16);       // tensor_dim0[15:0]  (bits 63:48)
  g1[2] = (int)(1u << 16);        // tensor_dim0[31:16]=0, tensor_dim1[15:0]=1
  g1[3] = (int)(ndw << 16);       // tensor_dim1[31:16]=0, tile_dim0=ndw
  g1[4] = 1;                      // tile_dim1=1, tile_dim2=0
  g1[5] = (int)ndw;               // tensor_dim0_stride[31:0]
  g1[6] = 0;                      // stride0[47:32]=0, stride1[15:0]=0
  g1[7] = 0;
  v4i_t z4 = {0, 0, 0, 0};
#if defined(__clang_major__) && __clang_major__ >= 23
  v8i_t z8 = {0, 0, 0, 0, 0, 0, 0, 0};
  __builtin_amdgcn_tensor_load_to_lds(g0, g1, z4, z4, z8, 0);
#else
  __builtin_amdgcn_tensor_load_to_lds(g0, g1, z4, z4, 0);
#endif
}
#endif  // USE_TDM

// ---------------------------------------------------------------------------
// Kernel 1: C(4096x512) = X(4096x4096) @ Wkv^T ; split into K(hi/lo) and V(bf16)
// Grid 256 (32 row-blocks x 8 col-blocks), block 256 (8 waves, 4x2 wave grid).
// ---------------------------------------------------------------------------
__global__ __launch_bounds__(256) void kv_proj_kernel(
    const float* __restrict__ X, const float* __restrict__ Wkv,
    unsigned short* __restrict__ khi_us, unsigned short* __restrict__ klo_us,
    unsigned short* __restrict__ vhi_us) {
  __bf16* __restrict__ Khi = reinterpret_cast<__bf16*>(khi_us);
  __bf16* __restrict__ Klo = reinterpret_cast<__bf16*>(klo_us);
  __bf16* __restrict__ Vhi = reinterpret_cast<__bf16*>(vhi_us);

  __shared__ __bf16 Xh[128 * 32], Xl[128 * 32], Wh[64 * 32], Wl[64 * 32];

  const int tid = threadIdx.x;
  const int lane = tid & 31;
  const int wave = tid >> 5;
  const int wr = wave >> 1;           // 0..3  (32-row strip)
  const int wc = wave & 1;            // 0..1  (32-col strip)
  const int rb = blockIdx.x >> 3;     // 0..31
  const int cb = blockIdx.x & 7;      // 0..7
  const int r0 = rb * 128;
  const int n0 = cb * 64;
  const int kbase = (lane < 16) ? 0 : 8;
  const int ln = lane & 15;
  const int half = (lane < 16) ? 0 : 8;

  v8f acc[2][2];
#pragma unroll
  for (int i = 0; i < 2; ++i)
#pragma unroll
    for (int j = 0; j < 2; ++j) acc[i][j] = vzero8();

  for (int k0 = 0; k0 < 4096; k0 += 32) {
    __syncthreads();
    // stage X 128x32 tile, splitting f32 -> bf16 hi + lo
#pragma unroll
    for (int it = 0; it < 16; ++it) {
      int idx = tid + it * 256;
      int r = idx >> 5, k = idx & 31;
      float v = X[(size_t)(r0 + r) * 4096 + k0 + k];
      __bf16 h = (__bf16)v;
      Xh[idx] = h;
      Xl[idx] = (__bf16)(v - (float)h);
    }
    // stage Wkv 64x32 tile (rows n0..n0+63)
#pragma unroll
    for (int it = 0; it < 8; ++it) {
      int idx = tid + it * 256;
      int r = idx >> 5, k = idx & 31;
      float v = Wkv[(size_t)(n0 + r) * 4096 + k0 + k];
      __bf16 h = (__bf16)v;
      Wh[idx] = h;
      Wl[idx] = (__bf16)(v - (float)h);
    }
    __syncthreads();

    v16bf ah[2], al[2], bh[2], bl[2];
#pragma unroll
    for (int i = 0; i < 2; ++i) {
      int row = wr * 32 + i * 16 + ln;
      ah[i] = frag_row(Xh + row * 32 + kbase);
      al[i] = frag_row(Xl + row * 32 + kbase);
    }
#pragma unroll
    for (int j = 0; j < 2; ++j) {
      int col = wc * 32 + j * 16 + ln;       // B = Wkv^T -> row-major read of Wkv
      bh[j] = frag_row(Wh + col * 32 + kbase);
      bl[j] = frag_row(Wl + col * 32 + kbase);
    }
#pragma unroll
    for (int i = 0; i < 2; ++i)
#pragma unroll
      for (int j = 0; j < 2; ++j) {
        acc[i][j] = wmma_bf16(ah[i], bh[j], acc[i][j]);  // hi*hi
        acc[i][j] = wmma_bf16(ah[i], bl[j], acc[i][j]);  // hi*lo
        acc[i][j] = wmma_bf16(al[i], bh[j], acc[i][j]);  // lo*hi
      }
  }

  // Epilogue: tile base is a multiple of 16 -> K/V choice is tile-uniform.
#pragma unroll
  for (int i = 0; i < 2; ++i)
#pragma unroll
    for (int j = 0; j < 2; ++j) {
      const int ctile = n0 + wc * 32 + j * 16;  // uniform across the wave
      if (ctile < 256) {
        const int colg = ctile + ln;
#pragma unroll
        for (int r = 0; r < 8; ++r) {
          int rowg = r0 + wr * 32 + i * 16 + r + half;
          float v = acc[i][j][r];
          __bf16 h = (__bf16)v;
          Khi[(size_t)rowg * 256 + colg] = h;
          Klo[(size_t)rowg * 256 + colg] = (__bf16)(v - (float)h);
        }
      } else {
        const int colg = ctile - 256 + ln;
#pragma unroll
        for (int r = 0; r < 8; ++r) {
          int rowg = r0 + wr * 32 + i * 16 + r + half;
          Vhi[(size_t)rowg * 256 + colg] = (__bf16)acc[i][j][r];
        }
      }
    }
}

// ---------------------------------------------------------------------------
// Kernel 2: flash attention. Grid 128, block 128 (4 waves x 16 query rows).
// AO(8192x256) f32 = softmax(Q K^T / 16, masked) @ V
// TDM double-buffer: DMA block n+1 while WMMAs run on block n.
// ---------------------------------------------------------------------------
__global__ __launch_bounds__(128) void attn_kernel(
    const float* __restrict__ Q, const int* __restrict__ mask,
    const unsigned short* __restrict__ khi_us,
    const unsigned short* __restrict__ klo_us,
    const unsigned short* __restrict__ vhi_us, float* __restrict__ AO) {
  const __bf16* __restrict__ Khi = reinterpret_cast<const __bf16*>(khi_us);
  const __bf16* __restrict__ Klo = reinterpret_cast<const __bf16*>(klo_us);
  const __bf16* __restrict__ Vhi = reinterpret_cast<const __bf16*>(vhi_us);

  __shared__ __bf16 Kh[2][32 * 256], Kl[2][32 * 256], Vh[2][32 * 256];  // 96 KB
  __shared__ __bf16 Pb[4][16 * 32];                                     // 4 KB

  const int tid = threadIdx.x;
  const int lane = tid & 31;
  const int wave = tid >> 5;
  const int kbase = (lane < 16) ? 0 : 8;
  const int ln = lane & 15;
  const int half = (lane < 16) ? 0 : 8;
  const int qrow0 = (blockIdx.x * 4 + wave) * 16;
  const float scale = 0.0625f;  // 1/sqrt(256)

  // Q fragments (pre-scaled), hi/lo split, held in registers: 8 k-chunks of 32
  v16bf qh[8], ql[8];
#pragma unroll
  for (int c = 0; c < 8; ++c) {
    const float* p = Q + (size_t)(qrow0 + ln) * 256 + c * 32 + kbase;
#pragma unroll
    for (int i = 0; i < 8; ++i) {
      float v = p[i] * scale;
      __bf16 h = (__bf16)v;
      qh[c][i] = h;
      ql[c][i] = (__bf16)(v - (float)h);
    }
#pragma unroll
    for (int i = 0; i < 8; ++i) {
      float v = p[16 + i] * scale;
      __bf16 h = (__bf16)v;
      qh[c][8 + i] = h;
      ql[c][8 + i] = (__bf16)(v - (float)h);
    }
  }

  v8f O[16];
#pragma unroll
  for (int t = 0; t < 16; ++t) O[t] = vzero8();
  float m[8], l[8];
#pragma unroll
  for (int r = 0; r < 8; ++r) {
    m[r] = -1e30f;
    l[r] = 0.f;
  }

#if USE_TDM
  // Prologue: DMA block 0 into buffer 0 (waves 0..2 own one stream each).
  if (wave == 0) {
    tdm_load_1d(Kh[0], Khi, 4096);
  } else if (wave == 1) {
    tdm_load_1d(Kl[0], Klo, 4096);
  } else if (wave == 2) {
    tdm_load_1d(Vh[0], Vhi, 4096);
  }
#endif

  for (int sb = 0; sb < 4096; sb += 32) {
    const int b = (sb >> 5) & 1;
#if USE_TDM
    const bool more = (sb + 32) < 4096;
    if (more) {  // issue next block into the other buffer
      const size_t goff = (size_t)(sb + 32) * 256;
      if (wave == 0) {
        tdm_load_1d(Kh[b ^ 1], Khi + goff, 4096);
      } else if (wave == 1) {
        tdm_load_1d(Kl[b ^ 1], Klo + goff, 4096);
      } else if (wave == 2) {
        tdm_load_1d(Vh[b ^ 1], Vhi + goff, 4096);
      }
    }
    if (wave < 3) {
      if (more)
        __builtin_amdgcn_s_wait_tensorcnt(1);  // current block done, next inflight
      else
        __builtin_amdgcn_s_wait_tensorcnt(0);
    }
    __syncthreads();  // current buffer visible to all waves
#else
    __syncthreads();
    {
      const uint32_t* gk = (const uint32_t*)(Khi + (size_t)sb * 256);
      const uint32_t* gl = (const uint32_t*)(Klo + (size_t)sb * 256);
      const uint32_t* gv = (const uint32_t*)(Vhi + (size_t)sb * 256);
      uint32_t* sk = (uint32_t*)Kh[b];
      uint32_t* sl = (uint32_t*)Kl[b];
      uint32_t* sv = (uint32_t*)Vh[b];
      for (int i = tid; i < 32 * 128; i += 128) {
        sk[i] = gk[i];
        sl[i] = gl[i];
        sv[i] = gv[i];
      }
    }
    __syncthreads();
#endif

    const __bf16* khp = Kh[b];
    const __bf16* klp = Kl[b];
    const __bf16* vhp = Vh[b];

    // scores tile (16 q-rows x 32 chunks) as two 16x16 accumulators
    v8f s0 = vzero8(), s1 = vzero8();
#pragma unroll
    for (int c = 0; c < 8; ++c) {
      const __bf16* ph0 = khp + (size_t)ln * 256 + c * 32 + kbase;
      const __bf16* pl0 = klp + (size_t)ln * 256 + c * 32 + kbase;
      v16bf k0h = frag_row(ph0), k0l = frag_row(pl0);
      s0 = wmma_bf16(qh[c], k0h, s0);
      s0 = wmma_bf16(qh[c], k0l, s0);
      s0 = wmma_bf16(ql[c], k0h, s0);
      const __bf16* ph1 = khp + (size_t)(16 + ln) * 256 + c * 32 + kbase;
      const __bf16* pl1 = klp + (size_t)(16 + ln) * 256 + c * 32 + kbase;
      v16bf k1h = frag_row(ph1), k1l = frag_row(pl1);
      s1 = wmma_bf16(qh[c], k1h, s1);
      s1 = wmma_bf16(qh[c], k1l, s1);
      s1 = wmma_bf16(ql[c], k1h, s1);
    }

    // mask -> -inf (lane owns one column per accumulator)
    if (mask[sb + ln] == 0) {
#pragma unroll
      for (int r = 0; r < 8; ++r) s0[r] = -1e30f;
    }
    if (mask[sb + 16 + ln] == 0) {
#pragma unroll
      for (int r = 0; r < 8; ++r) s1[r] = -1e30f;
    }

    // online softmax: row reductions across the 16 lanes holding each row
    float rm[8];
#pragma unroll
    for (int r = 0; r < 8; ++r) rm[r] = fmaxf(s0[r], s1[r]);
#pragma unroll
    for (int off = 1; off < 16; off <<= 1) {
#pragma unroll
      for (int r = 0; r < 8; ++r) rm[r] = fmaxf(rm[r], __shfl_xor(rm[r], off, 32));
    }
    float fac[8];
#pragma unroll
    for (int r = 0; r < 8; ++r) {
      float mn = fmaxf(m[r], rm[r]);
      fac[r] = __expf(m[r] - mn);
      m[r] = mn;
    }
#pragma unroll
    for (int r = 0; r < 8; ++r) {
      s0[r] = __expf(s0[r] - m[r]);
      s1[r] = __expf(s1[r] - m[r]);
    }
    float ps[8];
#pragma unroll
    for (int r = 0; r < 8; ++r) ps[r] = s0[r] + s1[r];
#pragma unroll
    for (int off = 1; off < 16; off <<= 1) {
#pragma unroll
      for (int r = 0; r < 8; ++r) ps[r] += __shfl_xor(ps[r], off, 32);
    }
#pragma unroll
    for (int r = 0; r < 8; ++r) l[r] = l[r] * fac[r] + ps[r];
#pragma unroll
    for (int t = 0; t < 16; ++t)
#pragma unroll
      for (int r = 0; r < 8; ++r) O[t][r] *= fac[r];

    // P (16x32) via per-wave LDS roundtrip, stored as bf16:
    // C-layout (lane=col, vgpr=row) -> A-fragment layout (lane=row)
    __bf16* pw = Pb[wave];
#pragma unroll
    for (int r = 0; r < 8; ++r) {
      pw[(r + half) * 32 + ln] = (__bf16)s0[r];
      pw[(r + half) * 32 + 16 + ln] = (__bf16)s1[r];
    }
    v16bf pf = frag_row(pw + ln * 32 + kbase);  // same-wave LDS: in-order

    // O += P @ V  (16 output column tiles)
#pragma unroll
    for (int t = 0; t < 16; ++t) {
      v16bf vf = frag_col(vhp + (size_t)kbase * 256 + t * 16 + ln, 256);
      O[t] = wmma_bf16(pf, vf, O[t]);
    }

#if USE_TDM
    __syncthreads();  // all reads of this buffer done before it is re-DMA'd
#endif
  }

  float inv[8];
#pragma unroll
  for (int r = 0; r < 8; ++r) inv[r] = (l[r] > 0.f) ? (1.0f / l[r]) : 0.f;
#pragma unroll
  for (int t = 0; t < 16; ++t) {
#pragma unroll
    for (int r = 0; r < 8; ++r) {
      AO[(size_t)(qrow0 + r + half) * 256 + t * 16 + ln] = O[t][r] * inv[r];
    }
  }
}

// ---------------------------------------------------------------------------
// Kernel 3: out = LN(AO @ Wout^T + Q). One block (256 thr) per row, 8192 rows.
// ---------------------------------------------------------------------------
__global__ __launch_bounds__(256) void proj_ln_kernel(
    const float* __restrict__ AO, const float* __restrict__ Q,
    const float* __restrict__ Wout, const float* __restrict__ gamma,
    const float* __restrict__ beta, float* __restrict__ out) {
  __shared__ float sa[256];
  __shared__ float red[256];
  const int row = blockIdx.x;
  const int t = threadIdx.x;

  sa[t] = AO[(size_t)row * 256 + t];
  __syncthreads();

  const float* w = Wout + (size_t)t * 256;
  float acc = 0.f;
#pragma unroll 8
  for (int k = 0; k < 256; ++k) acc += sa[k] * w[k];
  float h = acc + Q[(size_t)row * 256 + t];

  red[t] = h;
  __syncthreads();
  for (int s = 128; s > 0; s >>= 1) {
    if (t < s) red[t] += red[t + s];
    __syncthreads();
  }
  float mu = red[0] * (1.0f / 256.0f);
  __syncthreads();
  float d = h - mu;
  red[t] = d * d;
  __syncthreads();
  for (int s = 128; s > 0; s >>= 1) {
    if (t < s) red[t] += red[t + s];
    __syncthreads();
  }
  float var = red[0] * (1.0f / 256.0f);
  float rs = rsqrtf(var + 1e-5f);
  out[(size_t)row * 256 + t] = d * rs * gamma[t] + beta[t];
}

// ---------------------------------------------------------------------------
extern "C" void kernel_launch(void* const* d_in, const int* in_sizes, int n_in,
                              void* d_out, int out_size, void* d_ws,
                              size_t ws_size, hipStream_t stream) {
  const float* q = (const float*)d_in[0];     // (64,128,256) f32
  const float* x = (const float*)d_in[1];     // (1,4096,4096) f32
  const int* mask = (const int*)d_in[2];      // (4096,) i32
  const float* wkv = (const float*)d_in[3];   // (512,4096) f32
  const float* wout = (const float*)d_in[4];  // (256,256) f32
  const float* g = (const float*)d_in[5];     // (256,)
  const float* b = (const float*)d_in[6];     // (256,)
  float* out = (float*)d_out;                 // (64,128,256) f32

  // workspace layout (bf16 buffers handled as u16 on host side)
  unsigned short* Khi = (unsigned short*)d_ws;           // 4096*256
  unsigned short* Klo = Khi + (size_t)4096 * 256;        // 4096*256
  unsigned short* Vhi = Klo + (size_t)4096 * 256;        // 4096*256
  float* AO = (float*)(Vhi + (size_t)4096 * 256);        // 8192*256 f32

  hipLaunchKernelGGL(kv_proj_kernel, dim3(256), dim3(256), 0, stream, x, wkv,
                     Khi, Klo, Vhi);
  hipLaunchKernelGGL(attn_kernel, dim3(128), dim3(128), 0, stream, q, mask,
                     Khi, Klo, Vhi, AO);
  hipLaunchKernelGGL(proj_ln_kernel, dim3(8192), dim3(256), 0, stream, AO, q,
                     wout, g, b, out);
}
